// Attention_29798483100335
// MI455X (gfx1250) — matrix operand order repaired and checked by reading, not compiled, not run on hardware
//
#include <hip/hip_runtime.h>

// ---------------------------------------------------------------------------
// MI455X (gfx1250, wave32) attention pipeline, all GEMMs on v_wmma_f32_16x16x32_f16
// ---------------------------------------------------------------------------

typedef __attribute__((ext_vector_type(16))) _Float16 v16h;
typedef __attribute__((ext_vector_type(8)))  float    v8f;
typedef __attribute__((ext_vector_type(4)))  float    v4f;
typedef __attribute__((ext_vector_type(4)))  _Float16 v4h;
typedef __attribute__((ext_vector_type(4)))  unsigned int v4u;

union Frag {            // one WMMA 16-bit operand (8 VGPRs) as two 16B chunks
    v16h h;
    v4u  u[2];
};

__device__ __forceinline__ v8f wmma_f16(v16h a, v16h b, v8f c) {
    return __builtin_amdgcn_wmma_f32_16x16x32_f16(false, a, false, b,
                                                  (short)0, c, false, false);
}

__device__ __forceinline__ float rmax16(float v) {
    v = fmaxf(v, __shfl_xor(v, 1, 16));
    v = fmaxf(v, __shfl_xor(v, 2, 16));
    v = fmaxf(v, __shfl_xor(v, 4, 16));
    v = fmaxf(v, __shfl_xor(v, 8, 16));
    return v;
}
__device__ __forceinline__ float rsum16(float v) {
    v += __shfl_xor(v, 1, 16);
    v += __shfl_xor(v, 2, 16);
    v += __shfl_xor(v, 4, 16);
    v += __shfl_xor(v, 8, 16);
    return v;
}

constexpr int kB   = 8;
constexpr int kN   = 1024;
constexpr int kDim = 1024;
constexpr int kH   = 16;
constexpr int kDh  = 64;

// ---------------------------------------------------------------------------
// 1) LayerNorm: x fp32 [8192,1024] -> xn fp16
// ---------------------------------------------------------------------------
__global__ __launch_bounds__(256) void ln_kernel(const float* __restrict__ x,
                                                 const float* __restrict__ gamma,
                                                 const float* __restrict__ beta,
                                                 _Float16* __restrict__ xn) {
    const int row = blockIdx.x;                       // 0..8191
    const int t   = threadIdx.x;
    const float* xr = x + (size_t)row * kDim;
    v4f v = ((const v4f*)xr)[t];                      // 256*4 = 1024
    float s  = v[0] + v[1] + v[2] + v[3];
    float ss = v[0]*v[0] + v[1]*v[1] + v[2]*v[2] + v[3]*v[3];
    for (int m = 1; m < 32; m <<= 1) {
        s  += __shfl_xor(s,  m, 32);
        ss += __shfl_xor(ss, m, 32);
    }
    __shared__ float red[2][8];
    const int wid = t >> 5, lane = t & 31;
    if (lane == 0) { red[0][wid] = s; red[1][wid] = ss; }
    __syncthreads();
    float ts = 0.f, tss = 0.f;
#pragma unroll
    for (int i = 0; i < 8; ++i) { ts += red[0][i]; tss += red[1][i]; }
    const float mu  = ts * (1.f / kDim);
    const float var = tss * (1.f / kDim) - mu * mu;
    const float inv = rsqrtf(var + 1e-5f);
    v4f g = ((const v4f*)gamma)[t];
    v4f b = ((const v4f*)beta)[t];
    v4h o;
#pragma unroll
    for (int i = 0; i < 4; ++i)
        o[i] = (_Float16)((v[i] - mu) * inv * g[i] + b[i]);
    ((v4h*)(xn + (size_t)row * kDim))[t] = o;
}

// ---------------------------------------------------------------------------
// 2a) cos/sin tables for freqs and freqs+bias   (N*DH = 65536 elements)
// ---------------------------------------------------------------------------
__global__ __launch_bounds__(256) void trig_kernel(const float* __restrict__ freqs,
                                                   const float* __restrict__ bias,
                                                   float* __restrict__ cq, float* __restrict__ sq,
                                                   float* __restrict__ ck, float* __restrict__ sk) {
    const int i = blockIdx.x * 256 + threadIdx.x;
    if (i >= kN * kDh) return;
    const float f  = freqs[i];
    const float fb = f + bias[i];
    cq[i] = cosf(f);  sq[i] = sinf(f);
    ck[i] = cosf(fb); sk[i] = sinf(fb);
}

// ---------------------------------------------------------------------------
// 2b) fp32 [R][C] -> fp16 transposed [C][R]  (weights -> coalesced B tiles)
// ---------------------------------------------------------------------------
__global__ __launch_bounds__(256) void transpose_cast_kernel(const float* __restrict__ in,
                                                             _Float16* __restrict__ out,
                                                             int R, int C) {
    const int idx = blockIdx.x * 256 + threadIdx.x;
    if (idx >= R * C) return;
    const int r = idx / C, c = idx - r * C;
    out[(size_t)c * R + r] = (_Float16)in[idx];
}

// ---------------------------------------------------------------------------
// 3) QKV GEMM: xn[8192,1024] x W[1024,3072] with polar epilogue.
//    Block tile 128x64, BK=32; 8 waves, each 32x32 (2x2 WMMA 16x16x32 tiles).
// ---------------------------------------------------------------------------
__global__ __launch_bounds__(256) void gemm_qkv_kernel(
        const _Float16* __restrict__ A,      // xn [8192][1024]
        const _Float16* __restrict__ WqkT,   // [2048][1024]
        const _Float16* __restrict__ WvT,    // [1024][1024]
        const float* __restrict__ cq, const float* __restrict__ sq,
        const float* __restrict__ ck, const float* __restrict__ sk,
        _Float16* __restrict__ qp,           // [B*H][N][128]
        _Float16* __restrict__ kp,           // [B*H][N][128]
        _Float16* __restrict__ vh)           // [B*H][N][64]
{
    __shared__ _Float16 As[128 * 40];        // +8 pad: conflict-free b128 frags
    __shared__ _Float16 Bs[64 * 40];

    const int t = threadIdx.x;
    const int lane = t & 31, wid = t >> 5;
    const int m = lane & 15, half = lane >> 4;
    const int m0 = (wid & 3) * 32, n0 = (wid >> 2) * 32;
    const int blockN = blockIdx.x * 64;      // 0..3071
    const int blockM = blockIdx.y * 128;     // 0..8191

    const _Float16* Bsrc = (blockN < 2048)
                         ? (WqkT + (size_t)blockN * 1024)
                         : (WvT + (size_t)(blockN - 2048) * 1024);
    v8f acc[2][2] = {};

    for (int k0 = 0; k0 < 1024; k0 += 32) {
        __syncthreads();
#pragma unroll
        for (int rep = 0; rep < 2; ++rep) {  // A tile: 128 rows x 4 chunks
            int idx = t + rep * 256;
            int r = idx >> 2, ch = idx & 3;
            *(v4u*)&As[r * 40 + ch * 8] =
                *(const v4u*)(A + (size_t)(blockM + r) * 1024 + k0 + ch * 8);
        }
        {                                    // B tile: 64 rows x 4 chunks
            int r = t >> 2, ch = t & 3;
            *(v4u*)&Bs[r * 40 + ch * 8] =
                *(const v4u*)(Bsrc + (size_t)r * 1024 + k0 + ch * 8);
        }
        if (k0 + 32 < 1024) {
            __builtin_prefetch(A + (size_t)(blockM + (t >> 1)) * 1024 + k0 + 32, 0, 1);
            __builtin_prefetch(Bsrc + (size_t)(t >> 2) * 1024 + k0 + 32, 0, 1);
        }
        __syncthreads();

        Frag a[2], b[2];
#pragma unroll
        for (int i = 0; i < 2; ++i) {        // A-frag: lane=M, 16 consecutive K
            int base = (m0 + i * 16 + m) * 40 + 16 * half;
            a[i].u[0] = *(const v4u*)&As[base];
            a[i].u[1] = *(const v4u*)&As[base + 8];
        }
#pragma unroll
        for (int j = 0; j < 2; ++j) {        // B-frag: lane=N, 16 consecutive K
            int base = (n0 + j * 16 + m) * 40 + 16 * half;
            b[j].u[0] = *(const v4u*)&Bs[base];
            b[j].u[1] = *(const v4u*)&Bs[base + 8];
        }
#pragma unroll
        for (int i = 0; i < 2; ++i)
#pragma unroll
            for (int j = 0; j < 2; ++j)
                acc[i][j] = wmma_f16(a[i].h, b[j].h, acc[i][j]);
    }

    // Epilogue: C layout -> VGPR r = row r+8*half, lane&15 = col.
#pragma unroll
    for (int i = 0; i < 2; ++i)
#pragma unroll
        for (int j = 0; j < 2; ++j)
#pragma unroll
            for (int r = 0; r < 8; ++r) {
                const int grow = blockM + m0 + i * 16 + r + 8 * half;
                const int gcol = blockN + n0 + j * 16 + m;
                const float av = acc[i][j][r];
                const int bb = grow >> 10, n = grow & 1023;
                if (gcol < 2048) {
                    const bool isq = gcol < 1024;
                    const int e = isq ? gcol : (gcol - 1024);
                    const int head = e >> 6, d = e & 63;
                    const float sp = (av > 20.f) ? av : log1pf(__expf(av));
                    const float c = (isq ? cq : ck)[n * 64 + d];
                    const float s = (isq ? sq : sk)[n * 64 + d];
                    _Float16* dst = (isq ? qp : kp) +
                                    ((size_t)(bb * 16 + head) * 1024 + n) * 128;
                    dst[d]      = (_Float16)(sp * c);
                    dst[64 + d] = (_Float16)(sp * s);
                } else {
                    const int e = gcol - 2048;
                    const int head = e >> 6, d = e & 63;
                    vh[((size_t)(bb * 16 + head) * 1024 + n) * 64 + d] = (_Float16)av;
                }
            }
}

// ---------------------------------------------------------------------------
// 4) Flash attention. Grid: 1024 blocks = (b,h) x 8 query blocks of 128.
//    Each of 8 waves owns 16 query rows; 32-key blocks staged in LDS.
// ---------------------------------------------------------------------------
__global__ __launch_bounds__(256) void attn_kernel(
        const _Float16* __restrict__ qp,    // [B*H][N][128]
        const _Float16* __restrict__ kp,    // [B*H][N][128]
        const _Float16* __restrict__ vp,    // [B*H][N][64]
        _Float16* __restrict__ attn_out)    // [B*N][1024]
{
    __shared__ _Float16 Ks[32 * 136];       // [key][d], pad 128->136
    __shared__ _Float16 Vs[64 * 40];        // [d][key], pad 32->40
    __shared__ _Float16 Ps[8][16 * 40];     // per-wave P tile, C->A relayout

    const int qblk = blockIdx.x & 7;
    const int bh   = blockIdx.x >> 3;       // b*16 + h
    const _Float16* q = qp + (size_t)bh * kN * 128;
    const _Float16* k = kp + (size_t)bh * kN * 128;
    const _Float16* v = vp + (size_t)bh * kN * 64;

    const int t = threadIdx.x;
    const int lane = t & 31, wid = t >> 5;
    const int m = lane & 15, half = lane >> 4;
    const int q0 = qblk * 128 + wid * 16;

    // Resident Q A-frags: 16 rows x 128 d = 4 frags of 16x32
    Frag Qf[4];
    {
        const _Float16* qrow = q + (size_t)(q0 + m) * 128;
#pragma unroll
        for (int c = 0; c < 4; ++c) {
            Qf[c].u[0] = *(const v4u*)(qrow + 32 * c + 16 * half);
            Qf[c].u[1] = *(const v4u*)(qrow + 32 * c + 16 * half + 8);
        }
    }

    float mrow[8], lrow[8];
#pragma unroll
    for (int r = 0; r < 8; ++r) { mrow[r] = -1e30f; lrow[r] = 0.f; }
    v8f accO[4] = {};
    _Float16* myP = &Ps[wid][0];

    for (int kb = 0; kb < 32; ++kb) {
        const int k0 = kb * 32;
        __syncthreads();
        // Stage K block: 32 x 128 halves = 512 x 16B
#pragma unroll
        for (int rep = 0; rep < 2; ++rep) {
            int idx = t + rep * 256;
            int row = idx >> 4, ch = idx & 15;
            *(v4u*)&Ks[row * 136 + ch * 8] =
                *(const v4u*)(k + (size_t)(k0 + row) * 128 + ch * 8);
        }
        // Stage V block transposed: Vs[d][key]
        {
            int row = t >> 3, ch = t & 7;
            union { v4u u; _Float16 h[8]; } vv;
            vv.u = *(const v4u*)(v + (size_t)(k0 + row) * 64 + ch * 8);
#pragma unroll
            for (int e = 0; e < 8; ++e)
                Vs[(ch * 8 + e) * 40 + row] = vv.h[e];
        }
        __syncthreads();

        // S = Q K^T for 2 key sub-tiles of 16 (K-dim 128 = 4 WMMA each)
        v8f s0 = {}, s1 = {};
#pragma unroll
        for (int c = 0; c < 4; ++c) {
            Frag kf;
            int base = m * 136 + 32 * c + 16 * half;
            kf.u[0] = *(const v4u*)&Ks[base];
            kf.u[1] = *(const v4u*)&Ks[base + 8];
            s0 = wmma_f16(Qf[c].h, kf.h, s0);
            base = (16 + m) * 136 + 32 * c + 16 * half;
            kf.u[0] = *(const v4u*)&Ks[base];
            kf.u[1] = *(const v4u*)&Ks[base + 8];
            s1 = wmma_f16(Qf[c].h, kf.h, s1);
        }

        // Online softmax (row = r + 8*half, 16-lane group reductions)
        float corr[8];
#pragma unroll
        for (int r = 0; r < 8; ++r) {
            const float a0 = s0[r] * 0.125f;          // DH^-0.5 = 1/8
            const float a1 = s1[r] * 0.125f;
            const float mx   = rmax16(fmaxf(a0, a1));
            const float mnew = fmaxf(mrow[r], mx);
            const float p0 = __expf(a0 - mnew);
            const float p1 = __expf(a1 - mnew);
            myP[(r + 8 * half) * 40 + m]      = (_Float16)p0;
            myP[(r + 8 * half) * 40 + 16 + m] = (_Float16)p1;
            const float rs = rsum16(p0 + p1);
            corr[r] = __expf(mrow[r] - mnew);
            lrow[r] = lrow[r] * corr[r] + rs;
            mrow[r] = mnew;
        }
#pragma unroll
        for (int t4 = 0; t4 < 4; ++t4)
#pragma unroll
            for (int r = 0; r < 8; ++r)
                accO[t4][r] *= corr[r];

        // P (16x32) as A-frag from wave-private LDS
        Frag pf;
        {
            const int pb = m * 40 + 16 * half;
            pf.u[0] = *(const v4u*)&myP[pb];
            pf.u[1] = *(const v4u*)&myP[pb + 8];
        }
        // O += P V  (4 output d-tiles of 16)
#pragma unroll
        for (int t4 = 0; t4 < 4; ++t4) {
            Frag vf;
            const int vb = (t4 * 16 + m) * 40 + 16 * half;
            vf.u[0] = *(const v4u*)&Vs[vb];
            vf.u[1] = *(const v4u*)&Vs[vb + 8];
            accO[t4] = wmma_f16(pf.h, vf.h, accO[t4]);
        }
    }

    // Normalize and write attention output [b*N+n][h*64+d] fp16
    const int b = bh >> 4, hh = bh & 15;
#pragma unroll
    for (int t4 = 0; t4 < 4; ++t4)
#pragma unroll
        for (int r = 0; r < 8; ++r) {
            const int n = q0 + r + 8 * half;
            const float o = accO[t4][r] / lrow[r];
            attn_out[((size_t)(b * 1024 + n)) * 1024 + hh * 64 + t4 * 16 + m] =
                (_Float16)o;
        }
}

// ---------------------------------------------------------------------------
// 5) Output projection GEMM + bias -> fp32 out
// ---------------------------------------------------------------------------
__global__ __launch_bounds__(256) void gemm_out_kernel(
        const _Float16* __restrict__ A,     // attn [8192][1024]
        const _Float16* __restrict__ WoT,   // [1024][1024] (transposed w_out)
        const float* __restrict__ b_out,
        float* __restrict__ out)            // [8192][1024] fp32
{
    __shared__ _Float16 As[128 * 40];
    __shared__ _Float16 Bs[64 * 40];

    const int t = threadIdx.x;
    const int lane = t & 31, wid = t >> 5;
    const int m = lane & 15, half = lane >> 4;
    const int m0 = (wid & 3) * 32, n0 = (wid >> 2) * 32;
    const int blockN = blockIdx.x * 64;
    const int blockM = blockIdx.y * 128;
    const _Float16* Bsrc = WoT + (size_t)blockN * 1024;

    v8f acc[2][2] = {};

    for (int k0 = 0; k0 < 1024; k0 += 32) {
        __syncthreads();
#pragma unroll
        for (int rep = 0; rep < 2; ++rep) {
            int idx = t + rep * 256;
            int r = idx >> 2, ch = idx & 3;
            *(v4u*)&As[r * 40 + ch * 8] =
                *(const v4u*)(A + (size_t)(blockM + r) * 1024 + k0 + ch * 8);
        }
        {
            int r = t >> 2, ch = t & 3;
            *(v4u*)&Bs[r * 40 + ch * 8] =
                *(const v4u*)(Bsrc + (size_t)r * 1024 + k0 + ch * 8);
        }
        if (k0 + 32 < 1024) {
            __builtin_prefetch(A + (size_t)(blockM + (t >> 1)) * 1024 + k0 + 32, 0, 1);
            __builtin_prefetch(Bsrc + (size_t)(t >> 2) * 1024 + k0 + 32, 0, 1);
        }
        __syncthreads();

        Frag a[2], b[2];
#pragma unroll
        for (int i = 0; i < 2; ++i) {
            int base = (m0 + i * 16 + m) * 40 + 16 * half;
            a[i].u[0] = *(const v4u*)&As[base];
            a[i].u[1] = *(const v4u*)&As[base + 8];
        }
#pragma unroll
        for (int j = 0; j < 2; ++j) {
            int base = (n0 + j * 16 + m) * 40 + 16 * half;
            b[j].u[0] = *(const v4u*)&Bs[base];
            b[j].u[1] = *(const v4u*)&Bs[base + 8];
        }
#pragma unroll
        for (int i = 0; i < 2; ++i)
#pragma unroll
            for (int j = 0; j < 2; ++j)
                acc[i][j] = wmma_f16(a[i].h, b[j].h, acc[i][j]);
    }

#pragma unroll
    for (int i = 0; i < 2; ++i)
#pragma unroll
        for (int j = 0; j < 2; ++j)
#pragma unroll
            for (int r = 0; r < 8; ++r) {
                const int grow = blockM + m0 + i * 16 + r + 8 * half;
                const int gcol = blockN + n0 + j * 16 + m;
                out[(size_t)grow * 1024 + gcol] = acc[i][j][r] + b_out[gcol];
            }
}

// ---------------------------------------------------------------------------
// Host launcher
// ---------------------------------------------------------------------------
extern "C" void kernel_launch(void* const* d_in, const int* in_sizes, int n_in,
                              void* d_out, int out_size, void* d_ws, size_t ws_size,
                              hipStream_t stream) {
    (void)in_sizes; (void)n_in; (void)out_size; (void)ws_size;
    const float* x     = (const float*)d_in[0];
    const float* freqs = (const float*)d_in[1];
    const float* bias  = (const float*)d_in[2];
    const float* ln_g  = (const float*)d_in[3];
    const float* ln_b  = (const float*)d_in[4];
    const float* w_qk  = (const float*)d_in[5];
    const float* w_v   = (const float*)d_in[6];
    const float* w_out = (const float*)d_in[7];
    const float* b_out = (const float*)d_in[8];
    float* out = (float*)d_out;

    char* ws = (char*)d_ws;
    size_t off = 0;
    auto take = [&](size_t bytes) -> char* {
        char* p = ws + off;
        off += (bytes + 255) & ~(size_t)255;
        return p;
    };
    _Float16* xn   = (_Float16*)take((size_t)8192 * 1024 * 2);   // 16 MB
    _Float16* wqkT = (_Float16*)take((size_t)2048 * 1024 * 2);   //  4 MB
    _Float16* wvT  = (_Float16*)take((size_t)1024 * 1024 * 2);   //  2 MB
    _Float16* woT  = (_Float16*)take((size_t)1024 * 1024 * 2);   //  2 MB
    _Float16* qp   = (_Float16*)take((size_t)128 * 1024 * 128 * 2); // 32 MB
    _Float16* kp   = (_Float16*)take((size_t)128 * 1024 * 128 * 2); // 32 MB
    _Float16* vh   = (_Float16*)take((size_t)128 * 1024 * 64 * 2);  // 16 MB
    _Float16* ah   = (_Float16*)take((size_t)8192 * 1024 * 2);   // 16 MB
    float* cq = (float*)take((size_t)65536 * 4);
    float* sq = (float*)take((size_t)65536 * 4);
    float* ck = (float*)take((size_t)65536 * 4);
    float* sk = (float*)take((size_t)65536 * 4);

    ln_kernel<<<8192, 256, 0, stream>>>(x, ln_g, ln_b, xn);
    trig_kernel<<<256, 256, 0, stream>>>(freqs, bias, cq, sq, ck, sk);
    transpose_cast_kernel<<<(2048 * 1024 + 255) / 256, 256, 0, stream>>>(w_qk, wqkT, 1024, 2048);
    transpose_cast_kernel<<<(1024 * 1024 + 255) / 256, 256, 0, stream>>>(w_v, wvT, 1024, 1024);
    transpose_cast_kernel<<<(1024 * 1024 + 255) / 256, 256, 0, stream>>>(w_out, woT, 1024, 1024);
    gemm_qkv_kernel<<<dim3(48, 64), 256, 0, stream>>>(xn, wqkT, wvT, cq, sq, ck, sk, qp, kp, vh);
    attn_kernel<<<1024, 256, 0, stream>>>(qp, kp, vh, ah);
    gemm_out_kernel<<<dim3(16, 64), 256, 0, stream>>>(ah, woT, b_out, out);
}
